// BinaryGRUCell_63651415327175
// MI455X (gfx1250) — matrix operand order repaired and checked by reading, not compiled, not run on hardware
//
#include <hip/hip_runtime.h>
#include <hip/hip_bf16.h>
#include <math.h>

typedef __attribute__((ext_vector_type(16))) _Float16 v16h;
typedef __attribute__((ext_vector_type(8)))  _Float16 v8h;
typedef __attribute__((ext_vector_type(8)))  float    v8f;
typedef __attribute__((ext_vector_type(4)))  float    v4f;

#define T_STEPS 256
#define IN_DIM  128
#define H_DIM   256
#define G_DIM   768          // 3*H
#define ROWS_WG 16
#define HPAD    (H_DIM + 8)  // +16B per row -> rotates LDS banks, kills conflicts

// ---------------- weight f32 -> f16 conversion ----------------
__global__ void cvt_f32_f16_kernel(const float* __restrict__ src,
                                   _Float16* __restrict__ dst, int n) {
  int i = blockIdx.x * blockDim.x + threadIdx.x;
  if (i < n) dst[i] = (_Float16)src[i];
}

// ---------------- fused persistent binary-GRU ----------------
// A-fragment (16x32 f16): lane L<16 row M=L holds K[b..b+15];
//                         lane L+16 same row holds K[b+16..b+31].
// B-fragment (32x16 f16): lane L<16 col N=L holds K 0..15; lane L+16 same col K 16..31
//   -> 32 contiguous bytes of a row-major weight row.
// C/D tile: lane L holds col N=L&15, VGPR m holds row M = m + 8*(L>>4).
__global__ __launch_bounds__(256) void gru_fused_kernel(
    const float* __restrict__ x,        // [B, T, IN] f32
    const _Float16* __restrict__ wih,   // [768,128] f16
    const _Float16* __restrict__ whh,   // [768,256] f16
    const float* __restrict__ b_ih,     // [768]
    const float* __restrict__ b_hh,     // [768]
    float* __restrict__ out)            // [B, 256]
{
  __shared__ _Float16 hbuf[ROWS_WG][HPAD];

  const int tid  = threadIdx.x;
  const int wave = tid >> 5;
  const int lane = tid & 31;
  const int lrow = lane & 15;
  const int lhi  = lane >> 4;          // 0 or 1 (half-wave)
  const int row0 = blockIdx.x * ROWS_WG;

  // h0 = 0
  for (int i = tid; i < ROWS_WG * HPAD; i += blockDim.x)
    (&hbuf[0][0])[i] = (_Float16)0.f;
  __syncthreads();

  const int j0 = wave * 2;             // two 16-wide column tiles per wave

  // biases are time-invariant: hoist per-lane values
  float bR[2], bZ[2], bNI[2], bNH[2];
  #pragma unroll
  for (int jj = 0; jj < 2; ++jj) {
    const int col = (j0 + jj) * 16 + lrow;
    bR[jj]  = b_ih[col] + b_hh[col];
    bZ[jj]  = b_ih[col + H_DIM] + b_hh[col + H_DIM];
    bNI[jj] = b_ih[col + 2 * H_DIM];
    bNH[jj] = b_hh[col + 2 * H_DIM];
  }

  const float* xbase = x + (size_t)(row0 + lrow) * (T_STEPS * IN_DIM);

  // Opaque *offset* (not pointer): prevents LICM from hoisting the
  // t-invariant weight B-fragment loads out of the 256-step loop (which
  // previously spilled 576 VGPRs of weights to scratch), while PRESERVING
  // pointer provenance so the loads lower to global_load_b128 (LOADcnt only)
  // rather than flat_load (LOADcnt+DScnt, couples with the LDS pipe).
  // Weights (~576KB) are L2-resident, so re-streaming per step is cheap.
  size_t wofs = 0;

  #pragma unroll 1
  for (int t = 0; t < T_STEPS; ++t) {
    asm volatile("" : "+s"(wofs));
    const _Float16* __restrict__ wihp = wih + wofs;
    const _Float16* __restrict__ whhp = whh + wofs;

    const float* xr = xbase + t * IN_DIM;
    // prefetch next timestep's x row
    if (t + 1 < T_STEPS) __builtin_prefetch(xbase + (t + 1) * IN_DIM, 0, 1);

    // ---- x_t A-fragments: K = 128 -> 4 chunks (f32 -> f16 in-register) ----
    v16h ax[4];
    #pragma unroll
    for (int c = 0; c < 4; ++c) {
      const float* p0 = xr + c * 32 + lhi * 16;       // elems 0..7
      const float* p1 = p0 + 8;                       // elems 8..15
      v4f q0 = *(const v4f*)(p0);
      v4f q1 = *(const v4f*)(p0 + 4);
      v4f q2 = *(const v4f*)(p1);
      v4f q3 = *(const v4f*)(p1 + 4);
      v16h a;
      #pragma unroll
      for (int i = 0; i < 4; ++i) {
        a[i]      = (_Float16)q0[i];
        a[4 + i]  = (_Float16)q1[i];
        a[8 + i]  = (_Float16)q2[i];
        a[12 + i] = (_Float16)q3[i];
      }
      ax[c] = a;
    }

    // ---- h A-fragments from LDS: K = 256 -> 8 chunks ----
    v16h ah[8];
    const _Float16* hr = &hbuf[lrow][0];
    #pragma unroll
    for (int c = 0; c < 8; ++c) {
      const _Float16* p0 = hr + c * 32 + lhi * 16;
      v8h lo = *(const v8h*)(p0);
      v8h hi = *(const v8h*)(p0 + 8);
      v16h a;
      #pragma unroll
      for (int i = 0; i < 8; ++i) { a[i] = lo[i]; a[8 + i] = hi[i]; }
      ah[c] = a;
    }

    _Float16 hsig[2][8];

    #pragma unroll
    for (int jj = 0; jj < 2; ++jj) {
      const int nb = (j0 + jj) * 16 + lrow;           // B-frag column == gate row
      v8f accR  = {0.f,0.f,0.f,0.f,0.f,0.f,0.f,0.f};
      v8f accZ  = {0.f,0.f,0.f,0.f,0.f,0.f,0.f,0.f};
      v8f accNI = {0.f,0.f,0.f,0.f,0.f,0.f,0.f,0.f};
      v8f accNH = {0.f,0.f,0.f,0.f,0.f,0.f,0.f,0.f};

      // gi = x_t @ W_ih^T  (K = 128)
      #pragma unroll
      for (int c = 0; c < 4; ++c) {
        const int k = c * 32 + lhi * 16;
        v16h bRg = *(const v16h*)(wihp + (size_t)nb * IN_DIM + k);
        v16h bZg = *(const v16h*)(wihp + (size_t)(nb + H_DIM) * IN_DIM + k);
        v16h bNg = *(const v16h*)(wihp + (size_t)(nb + 2 * H_DIM) * IN_DIM + k);
        accR  = __builtin_amdgcn_wmma_f32_16x16x32_f16(false, ax[c], false, bRg, (short)0, accR,  false, false);
        accZ  = __builtin_amdgcn_wmma_f32_16x16x32_f16(false, ax[c], false, bZg, (short)0, accZ,  false, false);
        accNI = __builtin_amdgcn_wmma_f32_16x16x32_f16(false, ax[c], false, bNg, (short)0, accNI, false, false);
      }
      // gh = h @ W_hh^T  (K = 256); r,z share accumulators with gi, n kept split
      #pragma unroll
      for (int c = 0; c < 8; ++c) {
        const int k = c * 32 + lhi * 16;
        v16h bRg = *(const v16h*)(whhp + (size_t)nb * H_DIM + k);
        v16h bZg = *(const v16h*)(whhp + (size_t)(nb + H_DIM) * H_DIM + k);
        v16h bNg = *(const v16h*)(whhp + (size_t)(nb + 2 * H_DIM) * H_DIM + k);
        accR  = __builtin_amdgcn_wmma_f32_16x16x32_f16(false, ah[c], false, bRg, (short)0, accR,  false, false);
        accZ  = __builtin_amdgcn_wmma_f32_16x16x32_f16(false, ah[c], false, bZg, (short)0, accZ,  false, false);
        accNH = __builtin_amdgcn_wmma_f32_16x16x32_f16(false, ah[c], false, bNg, (short)0, accNH, false, false);
      }

      // gate math on the C tile (lane -> N = lrow; VGPR m -> M = m + 8*lhi)
      const int col = nb;
      #pragma unroll
      for (int m = 0; m < 8; ++m) {
        const int rowm = m + 8 * lhi;
        const float hold = (float)hbuf[rowm][col];
        const float r  = 1.f / (1.f + __expf(-(accR[m] + bR[jj])));
        const float z  = 1.f / (1.f + __expf(-(accZ[m] + bZ[jj])));
        const float nn = tanhf(accNI[m] + bNI[jj] + r * (accNH[m] + bNH[jj]));
        const float hv = (1.f - z) * nn + z * hold;
        hsig[jj][m] = (_Float16)((hv > 0.f) ? 1.f : ((hv < 0.f) ? -1.f : 0.f));
      }
    }

    __syncthreads();   // all waves finished reading old h
    #pragma unroll
    for (int jj = 0; jj < 2; ++jj) {
      const int col = (j0 + jj) * 16 + lrow;
      #pragma unroll
      for (int m = 0; m < 8; ++m) hbuf[m + 8 * lhi][col] = hsig[jj][m];
    }
    __syncthreads();   // new h visible to all waves
  }

  // write h_final (the sign values) as f32
  for (int i = tid; i < ROWS_WG * H_DIM; i += blockDim.x) {
    const int r = i >> 8;
    const int c = i & (H_DIM - 1);
    out[(size_t)(row0 + r) * H_DIM + c] = (float)hbuf[r][c];
  }
}

extern "C" void kernel_launch(void* const* d_in, const int* in_sizes, int n_in,
                              void* d_out, int out_size, void* d_ws, size_t ws_size,
                              hipStream_t stream) {
  const float* x    = (const float*)d_in[0];
  const float* W_ih = (const float*)d_in[1];
  const float* W_hh = (const float*)d_in[2];
  const float* b_ih = (const float*)d_in[3];
  const float* b_hh = (const float*)d_in[4];
  // d_in[5] = time_steps (== 256, baked into T_STEPS)

  const int B = in_sizes[0] / (T_STEPS * IN_DIM);   // 2048

  _Float16* wih16 = (_Float16*)d_ws;                         // 768*128 halves
  _Float16* whh16 = wih16 + (size_t)G_DIM * IN_DIM;          // 768*256 halves

  const int n1 = G_DIM * IN_DIM;
  const int n2 = G_DIM * H_DIM;
  cvt_f32_f16_kernel<<<(n1 + 255) / 256, 256, 0, stream>>>(W_ih, wih16, n1);
  cvt_f32_f16_kernel<<<(n2 + 255) / 256, 256, 0, stream>>>(W_hh, whh16, n2);

  gru_fused_kernel<<<B / ROWS_WG, 256, 0, stream>>>(x, wih16, whh16, b_ih, b_hh,
                                                    (float*)d_out);
}
